// FluxTransformerBlock_28853590294599
// MI455X (gfx1250) — compile-verified, matrix operand order checked by reading, not run on hardware
//
#include <hip/hip_runtime.h>
#include <hip/hip_bf16.h>

// ---------------------------------------------------------------------------
// FluxTransformerBlock for MI455X (gfx1250, wave32, WMMA bf16 + async LDS fill)
// ---------------------------------------------------------------------------
#define DIM   3072
#define NH_   24
#define DH_   128
#define INNER 3072
#define LI_   4096
#define LT_   512
#define LL    (LI_ + LT_)   // 4608
#define MLP_  12288
#define EPS_  1e-6f

typedef __bf16 bf16;
typedef __attribute__((ext_vector_type(16))) __bf16 v16bf;
typedef __attribute__((ext_vector_type(8)))  __bf16 v8bf;
typedef __attribute__((ext_vector_type(8)))  float  v8f;

__device__ __forceinline__ bf16 f2bf(float f) {
    unsigned u = __builtin_bit_cast(unsigned, f);
    unsigned r = u + 0x7FFFu + ((u >> 16) & 1u);   // round-to-nearest-even
    unsigned short s = (unsigned short)(r >> 16);
    return __builtin_bit_cast(bf16, s);
}
__device__ __forceinline__ float b2f(bf16 b) {
    unsigned short s = __builtin_bit_cast(unsigned short, b);
    unsigned u = ((unsigned)s) << 16;
    return __builtin_bit_cast(float, u);
}
__device__ __forceinline__ v8f zero8() {
    v8f z;
#pragma unroll
    for (int i = 0; i < 8; i++) z[i] = 0.f;
    return z;
}
// Build a 16x32 bf16 A/B fragment: halves [0..7] at p, [8..15] at p+16
// (CDNA5 wave32 layout: lane-half selects K offset 8*hi, upper K block +16)
__device__ __forceinline__ v16bf make_frag(const bf16* p) {
    v8bf lo = *(const v8bf*)p;
    v8bf hv = *(const v8bf*)(p + 16);
    v16bf a;
#pragma unroll
    for (int i = 0; i < 8; i++) { a[i] = lo[i]; a[i + 8] = hv[i]; }
    return a;
}
__device__ __forceinline__ v8f wmma_bf(v16bf a, v16bf b, v8f c) {
    return __builtin_amdgcn_wmma_f32_16x16x32_bf16(false, a, false, b, (short)0, c,
                                                   false, false);
}
__device__ __forceinline__ float gelu_tanh(float x) {
    float x3 = x * x * x;
    return 0.5f * x * (1.f + tanhf(0.7978845608028654f * (x + 0.044715f * x3)));
}

// --- CDNA5 async global->LDS copy (16B per lane), tracked by ASYNCcnt -------
__device__ __forceinline__ void async_load_b128(const bf16* gptr, void* lptr) {
    unsigned lds_off = (unsigned)(size_t)lptr;   // low 32 bits = LDS byte address
    asm volatile("global_load_async_to_lds_b128 %0, %1, off"
                 :: "v"(lds_off), "v"(gptr) : "memory");
}
__device__ __forceinline__ void wait_async0() {
#if __has_builtin(__builtin_amdgcn_s_wait_asynccnt)
    __builtin_amdgcn_s_wait_asynccnt(0);
#else
    asm volatile("s_wait_asynccnt 0x0" ::: "memory");
#endif
}

// ---------------------------------------------------------------------------
// 1) adaLN embedding: e = silu(temb) @ W_ada + b   (1x3072 @ 3072x18432), x2
// ---------------------------------------------------------------------------
__global__ __launch_bounds__(256) void ada_gemv(
    const float* __restrict__ temb,
    const float* __restrict__ Wimg, const float* __restrict__ bimg,
    const float* __restrict__ Wtxt, const float* __restrict__ btxt,
    float* __restrict__ eimg, float* __restrict__ etxt)
{
    const float* W = blockIdx.y ? Wtxt : Wimg;
    const float* bb = blockIdx.y ? btxt : bimg;
    float* out = blockIdx.y ? etxt : eimg;
    __shared__ float st[DIM];
    for (int i = threadIdx.x; i < DIM; i += 256) {
        float x = temb[i];
        st[i] = x / (1.f + __expf(-x));    // silu
    }
    __syncthreads();
    int col = blockIdx.x * 256 + threadIdx.x;
    float acc = bb[col];
#pragma unroll 8
    for (int k = 0; k < DIM; k++) acc += st[k] * W[(size_t)k * (6 * DIM) + col];
    out[col] = acc;
}

// ---------------------------------------------------------------------------
// 2) LayerNorm + modulate:  out = ln(x)*(1 + e[scaleOff+j]) + e[shiftOff+j]
// ---------------------------------------------------------------------------
__global__ __launch_bounds__(256) void ln_mod(
    const float* __restrict__ x, const float* __restrict__ e,
    int shiftOff, int scaleOff, bf16* __restrict__ out)
{
    const int row = blockIdx.x, t = threadIdx.x;
    const int lane = t & 31, w = t >> 5;
    const float* xr = x + (size_t)row * DIM;
    float v[12];
    float s = 0.f;
#pragma unroll
    for (int i = 0; i < 12; i++) { v[i] = xr[t + i * 256]; s += v[i]; }
    __shared__ float sm[8];
#pragma unroll
    for (int m = 1; m < 32; m <<= 1) s += __shfl_xor(s, m, 32);
    if (lane == 0) sm[w] = s;
    __syncthreads();
    s = 0.f;
#pragma unroll
    for (int j = 0; j < 8; j++) s += sm[j];
    const float mean = s * (1.f / DIM);
    __syncthreads();
    float s2 = 0.f;
#pragma unroll
    for (int i = 0; i < 12; i++) { float d = v[i] - mean; s2 += d * d; }
#pragma unroll
    for (int m = 1; m < 32; m <<= 1) s2 += __shfl_xor(s2, m, 32);
    if (lane == 0) sm[w] = s2;
    __syncthreads();
    s2 = 0.f;
#pragma unroll
    for (int j = 0; j < 8; j++) s2 += sm[j];
    const float rinv = rsqrtf(s2 * (1.f / DIM) + EPS_);
    bf16* orow = out + (size_t)row * DIM;
#pragma unroll
    for (int i = 0; i < 12; i++) {
        int col = t + i * 256;
        float lnv = (v[i] - mean) * rinv;
        orow[col] = f2bf(lnv * (1.f + e[scaleOff + col]) + e[shiftOff + col]);
    }
}

// ---------------------------------------------------------------------------
// 3) WMMA GEMM: C(MxN,f32acc) = A(bf16, MxK) * B(f32 global, KxN)
//    MODE 0: out bf16 = acc
//    MODE 1: out bf16 = gelu(acc + bias[n])
//    MODE 2: out f32  = resid[m,n] + gate[n]*(acc + bias[n])
//    Block 256 thr = 8 waves, tile 128x128, BK=32, wave tile 32x64.
//    A tile staged via async global->LDS b128; B tile converted f32->bf16.
// ---------------------------------------------------------------------------
template <int MODE>
__global__ __launch_bounds__(256) void gemm_bf16(
    const bf16* __restrict__ A, const float* __restrict__ B,
    int M, int N, int K,
    const float* __restrict__ bias, const float* __restrict__ gate,
    const float* __restrict__ resid,
    bf16* __restrict__ outb, float* __restrict__ outf)
{
    __shared__ __align__(16) bf16 lA[128 * 40];   // [m][k] pad-40
    __shared__ __align__(16) bf16 lB[128 * 40];   // [n][k] pad-40 (transposed)
    const int t = threadIdx.x;
    const int lane = t & 31, wid = t >> 5;
    const int hi = lane >> 4, ln = lane & 15;
    const int wm = wid & 3, wn = wid >> 2;         // 4 x 2 wave grid
    const int m0 = blockIdx.y * 128, n0 = blockIdx.x * 128;

    v8f acc[2][4];
#pragma unroll
    for (int i = 0; i < 2; i++)
#pragma unroll
        for (int j = 0; j < 4; j++) acc[i][j] = zero8();

    for (int kb = 0; kb < K; kb += 32) {
        // stage A: 128x32 bf16, async 16B chunks (no VGPR round-trip)
#pragma unroll
        for (int i = 0; i < 2; i++) {
            int c = t + i * 256;                   // 0..511
            int row = c >> 2, col = (c & 3) * 8;
            async_load_b128(A + (size_t)(m0 + row) * K + kb + col,
                            &lA[row * 40 + col]);
        }
        // stage B: 32x128 f32 -> bf16 transposed into lB[n][k]
#pragma unroll
        for (int i = 0; i < 4; i++) {
            int c = t + i * 256;                   // 0..1023
            int kr = c >> 5, n4 = (c & 31) * 4;
            float4 d = *(const float4*)(B + (size_t)(kb + kr) * N + n0 + n4);
            lB[(n4 + 0) * 40 + kr] = f2bf(d.x);
            lB[(n4 + 1) * 40 + kr] = f2bf(d.y);
            lB[(n4 + 2) * 40 + kr] = f2bf(d.z);
            lB[(n4 + 3) * 40 + kr] = f2bf(d.w);
        }
        wait_async0();
        __syncthreads();

        v16bf af[2], bfv[4];
#pragma unroll
        for (int mt = 0; mt < 2; mt++)
            af[mt] = make_frag(&lA[(wm * 32 + mt * 16 + ln) * 40 + 8 * hi]);
#pragma unroll
        for (int nt = 0; nt < 4; nt++)
            bfv[nt] = make_frag(&lB[(wn * 64 + nt * 16 + ln) * 40 + 8 * hi]);
#pragma unroll
        for (int mt = 0; mt < 2; mt++)
#pragma unroll
            for (int nt = 0; nt < 4; nt++)
                acc[mt][nt] = wmma_bf(af[mt], bfv[nt], acc[mt][nt]);
        __syncthreads();
    }

#pragma unroll
    for (int mt = 0; mt < 2; mt++)
#pragma unroll
        for (int nt = 0; nt < 4; nt++)
#pragma unroll
            for (int r = 0; r < 8; r++) {
                int row = m0 + wm * 32 + mt * 16 + r + 8 * hi;
                int col = n0 + wn * 64 + nt * 16 + ln;
                float v = acc[mt][nt][r];
                if (MODE == 0) {
                    outb[(size_t)row * N + col] = f2bf(v);
                } else if (MODE == 1) {
                    v += bias[col];
                    outb[(size_t)row * N + col] = f2bf(gelu_tanh(v));
                } else {
                    v += bias[col];
                    outf[(size_t)row * N + col] =
                        resid[(size_t)row * N + col] + gate[col] * v;
                }
            }
}

// ---------------------------------------------------------------------------
// 4) QKV fixup: RMS-norm (q,k) with scales, RoPE at merged positions, scatter
//    qkv_raw[row][9216] -> Q/K/V [h][pos][128] bf16   (pos: txt 0..511, img 512..)
// ---------------------------------------------------------------------------
__global__ __launch_bounds__(128) void qkv_fix(
    const bf16* __restrict__ raw, int is_txt,
    const float* __restrict__ qs, const float* __restrict__ ks,
    const float* __restrict__ rope,
    bf16* __restrict__ Q, bf16* __restrict__ K, bf16* __restrict__ V)
{
    const int row = blockIdx.x, h = blockIdx.y, d = threadIdx.x;
    const int lane = d & 31, w = d >> 5;
    const bf16* base = raw + (size_t)row * (3 * INNER) + h * DH_ + d;
    float q = b2f(base[0]);
    float k = b2f(base[INNER]);
    float v = b2f(base[2 * INNER]);
    const int pos = is_txt ? row : (LT_ + row);

    __shared__ float red[4];
    // q rms
    float ss = q * q;
#pragma unroll
    for (int m = 1; m < 32; m <<= 1) ss += __shfl_xor(ss, m, 32);
    if (lane == 0) red[w] = ss;
    __syncthreads();
    ss = red[0] + red[1] + red[2] + red[3];
    float qn = q * rsqrtf(ss * (1.f / DH_) + EPS_) * qs[d];
    __syncthreads();
    // k rms
    float sk = k * k;
#pragma unroll
    for (int m = 1; m < 32; m <<= 1) sk += __shfl_xor(sk, m, 32);
    if (lane == 0) red[w] = sk;
    __syncthreads();
    sk = red[0] + red[1] + red[2] + red[3];
    float kn = k * rsqrtf(sk * (1.f / DH_) + EPS_) * ks[d];

    // rope: fr[pos][d/2][c][j] at pos*256 + (d/2)*4 + c*2 + j ; c = d&1
    const float* frp = rope + (size_t)pos * 256 + (d >> 1) * 4;
    float f0 = (d & 1) ? frp[2] : frp[0];
    float f1 = (d & 1) ? frp[3] : frp[1];
    {
        float other = __shfl_xor(qn, 1, 32);
        float x0 = (d & 1) ? other : qn;
        float x1 = (d & 1) ? qn : other;
        qn = f0 * x0 + f1 * x1;
    }
    {
        float other = __shfl_xor(kn, 1, 32);
        float x0 = (d & 1) ? other : kn;
        float x1 = (d & 1) ? kn : other;
        kn = f0 * x0 + f1 * x1;
    }
    size_t idx = ((size_t)h * LL + pos) * DH_ + d;
    Q[idx] = f2bf(qn);
    K[idx] = f2bf(kn);
    V[idx] = f2bf(v);
}

// ---------------------------------------------------------------------------
// 5) Flash attention, one head per blockIdx.y, 128 q rows per blockIdx.x.
//    8 waves; each wave owns 16 q rows; Q fragments resident in VGPRs.
//    S = Q*K^T and O += P*V via v_wmma_f32_16x16x32_bf16.
//    K tile filled with async global->LDS b128 copies.
// ---------------------------------------------------------------------------
__global__ __launch_bounds__(256) void attn_kernel(
    const bf16* __restrict__ Qg, const bf16* __restrict__ Kg,
    const bf16* __restrict__ Vg, bf16* __restrict__ outb)
{
    const int h = blockIdx.y, qt = blockIdx.x;
    const int t = threadIdx.x, lane = t & 31, wid = t >> 5;
    const int hi = lane >> 4, ln = lane & 15;
    const bf16* Qh = Qg + (size_t)h * LL * DH_;
    const bf16* Kh = Kg + (size_t)h * LL * DH_;
    const bf16* Vh = Vg + (size_t)h * LL * DH_;

    __shared__ __align__(16) bf16 lK[64 * 136];       // [j][d] pad-136
    __shared__ __align__(16) bf16 lV[128 * 72];       // [d][j] pad-72 (transposed)
    __shared__ __align__(16) bf16 lP[8 * 16 * 72];    // per-wave [m][j] pad-72

    const int qrow0 = qt * 128 + wid * 16;
    v16bf aq[4];
#pragma unroll
    for (int c = 0; c < 4; c++)
        aq[c] = make_frag(Qh + (size_t)(qrow0 + ln) * DH_ + c * 32 + 8 * hi);

    v8f o[8];
#pragma unroll
    for (int i = 0; i < 8; i++) o[i] = zero8();
    float mrow[8], lrow[8];
#pragma unroll
    for (int r = 0; r < 8; r++) { mrow[r] = -1e30f; lrow[r] = 0.f; }
    const float sc = 0.08838834764831845f;            // 1/sqrt(128)

    for (int j0 = 0; j0 < LL; j0 += 64) {
        // stage K tile (row major) via async copies
#pragma unroll
        for (int i = 0; i < 2; i++) {
            int c = t + i * 256;                      // 0..511 chunks of 16
            int jr = c >> 3, d0 = (c & 7) * 16;
            const bf16* g = Kh + (size_t)(j0 + jr) * DH_ + d0;
            async_load_b128(g, &lK[jr * 136 + d0]);
            async_load_b128(g + 8, &lK[jr * 136 + d0 + 8]);
        }
        // stage V tile transposed (VALU path)
#pragma unroll
        for (int i = 0; i < 4; i++) {
            int c = t + i * 256;                      // 0..1023 chunks of 8
            int jr = c >> 4, d0 = (c & 15) * 8;
            v8bf dv = *(const v8bf*)(Vh + (size_t)(j0 + jr) * DH_ + d0);
#pragma unroll
            for (int e = 0; e < 8; e++) lV[(d0 + e) * 72 + jr] = dv[e];
        }
        wait_async0();
        __syncthreads();

        // S = Q * K^T   (16x64 per wave)
        v8f s[4];
#pragma unroll
        for (int jt = 0; jt < 4; jt++) s[jt] = zero8();
#pragma unroll
        for (int jt = 0; jt < 4; jt++)
#pragma unroll
            for (int c = 0; c < 4; c++) {
                v16bf b = make_frag(&lK[(jt * 16 + ln) * 136 + c * 32 + 8 * hi]);
                s[jt] = wmma_bf(aq[c], b, s[jt]);
            }

        // online softmax update
        float cf[8];
#pragma unroll
        for (int r = 0; r < 8; r++) {
            float mx = s[0][r];
#pragma unroll
            for (int jt = 1; jt < 4; jt++) mx = fmaxf(mx, s[jt][r]);
#pragma unroll
            for (int m = 1; m < 16; m <<= 1) mx = fmaxf(mx, __shfl_xor(mx, m, 16));
            float mnew = fmaxf(mrow[r], sc * mx);
            cf[r] = __expf(mrow[r] - mnew);
            mrow[r] = mnew;
            float rs = 0.f;
#pragma unroll
            for (int jt = 0; jt < 4; jt++) {
                float p = __expf(sc * s[jt][r] - mnew);
                s[jt][r] = p;
                rs += p;
            }
#pragma unroll
            for (int m = 1; m < 16; m <<= 1) rs += __shfl_xor(rs, m, 16);
            lrow[r] = lrow[r] * cf[r] + rs;
        }
#pragma unroll
        for (int dt = 0; dt < 8; dt++)
#pragma unroll
            for (int r = 0; r < 8; r++) o[dt][r] *= cf[r];

        // P: D-layout -> A-layout via per-wave LDS scratch
        bf16* pw = &lP[wid * 16 * 72];
#pragma unroll
        for (int jt = 0; jt < 4; jt++)
#pragma unroll
            for (int r = 0; r < 8; r++)
                pw[(r + 8 * hi) * 72 + jt * 16 + ln] = f2bf(s[jt][r]);
        v16bf ap[2];
#pragma unroll
        for (int c2 = 0; c2 < 2; c2++)
            ap[c2] = make_frag(&pw[ln * 72 + c2 * 32 + 8 * hi]);

        // O += P * V
#pragma unroll
        for (int dt = 0; dt < 8; dt++)
#pragma unroll
            for (int c2 = 0; c2 < 2; c2++) {
                v16bf b = make_frag(&lV[(dt * 16 + ln) * 72 + c2 * 32 + 8 * hi]);
                o[dt] = wmma_bf(ap[c2], b, o[dt]);
            }
        __syncthreads();
    }

    // finalize: out[pos][h*128 + d] bf16
#pragma unroll
    for (int r = 0; r < 8; r++) {
        float inv = 1.f / lrow[r];
        int row = qt * 128 + wid * 16 + r + 8 * hi;
#pragma unroll
        for (int dt = 0; dt < 8; dt++) {
            int col = h * DH_ + dt * 16 + ln;
            outb[(size_t)row * INNER + col] = f2bf(o[dt][r] * inv);
        }
    }
}

// ---------------------------------------------------------------------------
// Launch
// ---------------------------------------------------------------------------
extern "C" void kernel_launch(void* const* d_in, const int* in_sizes, int n_in,
                              void* d_out, int out_size, void* d_ws, size_t ws_size,
                              hipStream_t stream)
{
    (void)in_sizes; (void)n_in; (void)out_size; (void)ws_size;
    const float* hidden = (const float*)d_in[0];
    const float* enc    = (const float*)d_in[1];
    const float* temb   = (const float*)d_in[2];
    const float* rope   = (const float*)d_in[3];
    const float* W_img_ada = (const float*)d_in[4];
    const float* b_img_ada = (const float*)d_in[5];
    const float* W_txt_ada = (const float*)d_in[6];
    const float* b_txt_ada = (const float*)d_in[7];
    const float* W_iqkv  = (const float*)d_in[8];
    const float* W_eqkv  = (const float*)d_in[9];
    const float* W_iproj = (const float*)d_in[10];
    const float* b_iproj = (const float*)d_in[11];
    const float* W_eproj = (const float*)d_in[12];
    const float* b_eproj = (const float*)d_in[13];
    const float* q_scale  = (const float*)d_in[14];
    const float* k_scale  = (const float*)d_in[15];
    const float* eq_scale = (const float*)d_in[16];
    const float* ek_scale = (const float*)d_in[17];
    const float* W_img_mlp1 = (const float*)d_in[18];
    const float* b_img_mlp1 = (const float*)d_in[19];
    const float* W_img_mlp2 = (const float*)d_in[20];
    const float* b_img_mlp2 = (const float*)d_in[21];
    const float* W_txt_mlp1 = (const float*)d_in[22];
    const float* b_txt_mlp1 = (const float*)d_in[23];
    const float* W_txt_mlp2 = (const float*)d_in[24];
    const float* b_txt_mlp2 = (const float*)d_in[25];

    char* ws = (char*)d_ws;
    // workspace layout (bytes, lifetime-aliased)
    const size_t OFF_EIMG = 0;                                  // 73728
    const size_t OFF_ETXT = 73728;                              // 73728
    const size_t OFF_HS   = 147456;                             // 50331648
    const size_t OFF_ES   = 50479104;                           // 6291456
    const size_t OFF_NH   = 56770560;                           // 25165824 (nh / nh2)
    const size_t OFF_NE   = 81936384;                           // 3145728  (ne / ne2)
    const size_t OFF_AR   = 85082112;                           // 113246208 arena
    const size_t OFF_QKVB = 198328320;                          // 84934656 Q/K/V

    float* e_img = (float*)(ws + OFF_EIMG);
    float* e_txt = (float*)(ws + OFF_ETXT);
    float* hs    = (float*)(ws + OFF_HS);
    float* es    = (float*)(ws + OFF_ES);
    bf16*  nh    = (bf16*)(ws + OFF_NH);
    bf16*  ne    = (bf16*)(ws + OFF_NE);
    // arena phase 1: qkv_raw (img, txt) + attn_out ; phase 2: ff (img, txt)
    bf16* qkv_img  = (bf16*)(ws + OFF_AR);
    bf16* qkv_txt  = (bf16*)(ws + OFF_AR + (size_t)LI_ * 3 * INNER * 2);
    bf16* attn_out = (bf16*)(ws + OFF_AR + (size_t)LL * 3 * INNER * 2);
    bf16* ff_img   = (bf16*)(ws + OFF_AR);
    bf16* ff_txt   = (bf16*)(ws + OFF_AR + (size_t)LI_ * MLP_ * 2);
    bf16* Qb = (bf16*)(ws + OFF_QKVB);
    bf16* Kb = Qb + (size_t)NH_ * LL * DH_;
    bf16* Vb = Kb + (size_t)NH_ * LL * DH_;
    float* out_hs = (float*)d_out;
    float* out_es = out_hs + (size_t)LI_ * DIM;

    // 1) adaLN embeddings
    ada_gemv<<<dim3(6 * DIM / 256, 2), 256, 0, stream>>>(
        temb, W_img_ada, b_img_ada, W_txt_ada, b_txt_ada, e_img, e_txt);
    // 2) LN+modulate (msa)
    ln_mod<<<LI_, 256, 0, stream>>>(hidden, e_img, 0, DIM, nh);
    ln_mod<<<LT_, 256, 0, stream>>>(enc, e_txt, 0, DIM, ne);
    // 3) QKV GEMMs
    gemm_bf16<0><<<dim3(72, 32), 256, 0, stream>>>(
        nh, W_iqkv, LI_, 3 * INNER, DIM, nullptr, nullptr, nullptr, qkv_img, nullptr);
    gemm_bf16<0><<<dim3(72, 4), 256, 0, stream>>>(
        ne, W_eqkv, LT_, 3 * INNER, DIM, nullptr, nullptr, nullptr, qkv_txt, nullptr);
    // 4) RMS + RoPE + scatter
    qkv_fix<<<dim3(LI_, NH_), 128, 0, stream>>>(
        qkv_img, 0, q_scale, k_scale, rope, Qb, Kb, Vb);
    qkv_fix<<<dim3(LT_, NH_), 128, 0, stream>>>(
        qkv_txt, 1, eq_scale, ek_scale, rope, Qb, Kb, Vb);
    // 5) Attention
    attn_kernel<<<dim3(LL / 128, NH_), 256, 0, stream>>>(Qb, Kb, Vb, attn_out);
    // 6) Output projections (+gate, +residual)
    gemm_bf16<2><<<dim3(24, 32), 256, 0, stream>>>(
        attn_out + (size_t)LT_ * INNER, W_iproj, LI_, DIM, INNER,
        b_iproj, e_img + 2 * DIM, hidden, nullptr, hs);
    gemm_bf16<2><<<dim3(24, 4), 256, 0, stream>>>(
        attn_out, W_eproj, LT_, DIM, INNER,
        b_eproj, e_txt + 2 * DIM, enc, nullptr, es);
    // 7) LN+modulate (mlp)
    ln_mod<<<LI_, 256, 0, stream>>>(hs, e_img, 3 * DIM, 4 * DIM, nh);
    ln_mod<<<LT_, 256, 0, stream>>>(es, e_txt, 3 * DIM, 4 * DIM, ne);
    // 8) MLPs
    gemm_bf16<1><<<dim3(96, 32), 256, 0, stream>>>(
        nh, W_img_mlp1, LI_, MLP_, DIM, b_img_mlp1, nullptr, nullptr, ff_img, nullptr);
    gemm_bf16<2><<<dim3(24, 32), 256, 0, stream>>>(
        ff_img, W_img_mlp2, LI_, DIM, MLP_, b_img_mlp2, e_img + 5 * DIM, hs,
        nullptr, out_hs);
    gemm_bf16<1><<<dim3(96, 4), 256, 0, stream>>>(
        ne, W_txt_mlp1, LT_, MLP_, DIM, b_txt_mlp1, nullptr, nullptr, ff_txt, nullptr);
    gemm_bf16<2><<<dim3(24, 4), 256, 0, stream>>>(
        ff_txt, W_txt_mlp2, LT_, DIM, MLP_, b_txt_mlp2, e_txt + 5 * DIM, es,
        nullptr, out_es);
}
